// Transformer_36266703847566
// MI455X (gfx1250) — matrix operand order repaired and checked by reading, not compiled, not run on hardware
//
#include <hip/hip_runtime.h>
#include <math.h>

// ---------------------------------------------------------------------------
// Transformer forward (12 layers) for gfx1250 (MI455X), wave32 + WMMA.
// GEMMs: f32 weights staged Global->LDS with GLOBAL_LOAD_ASYNC_TO_LDS_B128
// (double-buffered, ASYNCcnt-tracked), converted to f16 in registers, and
// consumed by v_wmma_f32_16x16x32_f16 with f32 accumulation. HBM-bound.
// ---------------------------------------------------------------------------

typedef __attribute__((ext_vector_type(16))) _Float16 v16h;
typedef __attribute__((ext_vector_type(8)))  float    v8f;

#define LAYERS 12
#define DMODEL 512
#define NHEAD  8
#define HEADD  64
#define FFDIM  2048
#define VOCAB  100256
#define BATCH  64
#define SEQ    16
#define MROWS  (BATCH*SEQ)      // 1024 token rows

// LDS slab layout (per buffer, dwords): rows k=0..15 at [k*32 + n],
// rows k=16..31 at [528 + (k-15-1)*32 + n]  (64B pad -> +16 bank shift,
// so lanes 0-15 and 16-31 of a wave read disjoint bank groups).
#define SLAB_HALF   528            // 16*32 + 16 pad dwords
#define SLAB_DWORDS 1040           // two halves (512 + 16 + 512)

// ---------------- CDNA5 async global->LDS helpers --------------------------
__device__ __forceinline__ void async_b128_to_lds(unsigned lds_byte_off, const void* gaddr) {
  asm volatile("global_load_async_to_lds_b128 %0, %1, off"
               :: "v"(lds_byte_off), "v"(gaddr) : "memory");
}
__device__ __forceinline__ void wait_async_le1() {
  asm volatile("s_wait_asynccnt 0x1" ::: "memory");
}
__device__ __forceinline__ void wait_async_le0() {
  asm volatile("s_wait_asynccnt 0x0" ::: "memory");
}

// ---------------- WMMA A fragment loader (f32 memory -> f16 regs) ----------
// A fragment: 16x32 (MxK). lane m = lane&15; lanes>=16 hold K offset +16.
// halves 0..15 -> K = k0+ka+0..15
__device__ __forceinline__ v16h load_a_frag_f32(const float* __restrict__ A,
                                                int lda, int row0, int k0, int lane) {
  const int m  = lane & 15;
  const int ka = (lane >> 4) << 4;                 // 0 or 16
  const float* p = A + (size_t)(row0 + m) * lda + k0 + ka;
  float4 a0 = *(const float4*)(p);
  float4 a1 = *(const float4*)(p + 4);
  float4 a2 = *(const float4*)(p + 8);
  float4 a3 = *(const float4*)(p + 12);
  v16h r;
  r[0]=(_Float16)a0.x;  r[1]=(_Float16)a0.y;  r[2]=(_Float16)a0.z;  r[3]=(_Float16)a0.w;
  r[4]=(_Float16)a1.x;  r[5]=(_Float16)a1.y;  r[6]=(_Float16)a1.z;  r[7]=(_Float16)a1.w;
  r[8]=(_Float16)a2.x;  r[9]=(_Float16)a2.y;  r[10]=(_Float16)a2.z; r[11]=(_Float16)a2.w;
  r[12]=(_Float16)a3.x; r[13]=(_Float16)a3.y; r[14]=(_Float16)a3.z; r[15]=(_Float16)a3.w;
  return r;
}

// ---------------- Generic GEMM: C[M,N] = A[M,K] @ W[K,N] (+epilogue) -------
// Block = 256 thr = 8 waves, macro-tile M64 x N32 (wave w: row (w>>1), col (w&1)).
// Requires: M % 64 == 0, N % 32 == 0, K % 32 == 0 (true for all uses).
// epi: 0 = none, 1 = +bias, 2 = relu(.+bias), 3 = +bias +resid
__global__ __launch_bounds__(256) void gpt_gemm_wmma(
    const float* __restrict__ A, const float* __restrict__ W,
    const float* __restrict__ bias, const float* __restrict__ resid,
    float* __restrict__ C, int M, int N, int K, int epi) {
  __shared__ float sbuf[2 * SLAB_DWORDS];

  const int lane = threadIdx.x & 31;
  const int wid  = threadIdx.x >> 5;
  const int nbn  = N >> 5;                          // blocks along N
  const int rowBase = (blockIdx.x / nbn) << 6;      // 64-row macro tile
  const int colBase = (blockIdx.x % nbn) << 5;      // 32-col macro tile
  const int row0 = rowBase + ((wid >> 1) << 4);
  const int col0 = colBase + ((wid & 1) << 4);

  // staging job for this thread: one 16B group of the 32x32 slab
  const int sk = threadIdx.x >> 3;                  // k row 0..31
  const int sn = (threadIdx.x & 7) << 2;            // n group 0,4,..,28
  const unsigned slab_off =
      (unsigned)(((sk & 15) * 32 + (sk >> 4) * SLAB_HALF + sn) * 4);
  const unsigned sbase = (unsigned)(size_t)&sbuf[0];
  const float* gsrc0 = W + (size_t)sk * N + colBase + sn;

  const int nk = K >> 5;
  // prologue: stage k-step 0 into buffer 0
  async_b128_to_lds(sbase + slab_off, gsrc0);

  // fragment read geometry (from LDS)
  const int nl   = (wid & 1) * 16 + (lane & 15);    // local column 0..31
  const int half = (lane >> 4) * SLAB_HALF;         // K 0..15 vs 16..31

  v8f acc = {};
  for (int ks = 0; ks < nk; ++ks) {
    const int cur = ks & 1;
    if (ks + 1 < nk) {
      async_b128_to_lds(sbase + (unsigned)((ks + 1) & 1) * (SLAB_DWORDS * 4) + slab_off,
                        gsrc0 + (size_t)(ks + 1) * 32 * N);
      wait_async_le1();                             // current slab landed
    } else {
      wait_async_le0();
    }
    __syncthreads();                                // slab visible to all waves

    v16h af = load_a_frag_f32(A, K, row0, ks << 5, lane);
    const float* lb = &sbuf[cur * SLAB_DWORDS + half + nl];
    v16h bf;
#pragma unroll
    for (int j = 0; j < 16; ++j) bf[j] = (_Float16)lb[j * 32];
    acc = __builtin_amdgcn_wmma_f32_16x16x32_f16(false, af, false, bf,
                                                 (short)0, acc, false, false);
    __syncthreads();                                // reads done before re-stage
  }

  const int n  = lane & 15;
  const int mo = (lane >> 4) << 3;
  const float bv = epi ? bias[col0 + n] : 0.0f;
#pragma unroll
  for (int r = 0; r < 8; ++r) {
    size_t idx = (size_t)(row0 + mo + r) * N + col0 + n;
    float v = acc[r] + bv;
    if (epi == 2) v = v > 0.0f ? v : 0.0f;
    if (epi == 3) v += resid[idx];
    C[idx] = v;
  }
}

// ---------------- Attention: scores = Q @ K^T (per batch,head) -------------
// q,k layout: [B, T, H, HD]. One wave per (b,h): 16x16 tile, K = HD = 64.
__global__ __launch_bounds__(64) void gpt_attn_scores(
    const float* __restrict__ q, const float* __restrict__ kmat,
    float* __restrict__ scores) {
  const int lane = threadIdx.x & 31;
  const int bh   = blockIdx.x * 2 + (threadIdx.x >> 5);  // 0 .. B*H-1
  const int b    = bh >> 3;
  const int hh   = bh & 7;
  const int mn   = lane & 15;
  const int ka   = (lane >> 4) << 3;
  const int kb   = (lane >> 4) << 3;

  v8f acc = {};
  for (int k0 = 0; k0 < HEADD; k0 += 32) {
    const float* pa = q + (((size_t)b * SEQ + mn) * NHEAD + hh) * HEADD + k0 + ka;
    float4 a0 = *(const float4*)(pa);
    float4 a1 = *(const float4*)(pa + 4);
    float4 a2 = *(const float4*)(pa + 16);
    float4 a3 = *(const float4*)(pa + 20);
    v16h af;
    af[0]=(_Float16)a0.x;  af[1]=(_Float16)a0.y;  af[2]=(_Float16)a0.z;  af[3]=(_Float16)a0.w;
    af[4]=(_Float16)a1.x;  af[5]=(_Float16)a1.y;  af[6]=(_Float16)a1.z;  af[7]=(_Float16)a1.w;
    af[8]=(_Float16)a2.x;  af[9]=(_Float16)a2.y;  af[10]=(_Float16)a2.z; af[11]=(_Float16)a2.w;
    af[12]=(_Float16)a3.x; af[13]=(_Float16)a3.y; af[14]=(_Float16)a3.z; af[15]=(_Float16)a3.w;
    const float* pb = kmat + (((size_t)b * SEQ + mn) * NHEAD + hh) * HEADD + k0 + kb;
    float4 b0 = *(const float4*)(pb);
    float4 b1 = *(const float4*)(pb + 4);
    float4 b2 = *(const float4*)(pb + 16);
    float4 b3 = *(const float4*)(pb + 20);
    v16h bf;
    bf[0]=(_Float16)b0.x;  bf[1]=(_Float16)b0.y;  bf[2]=(_Float16)b0.z;  bf[3]=(_Float16)b0.w;
    bf[4]=(_Float16)b1.x;  bf[5]=(_Float16)b1.y;  bf[6]=(_Float16)b1.z;  bf[7]=(_Float16)b1.w;
    bf[8]=(_Float16)b2.x;  bf[9]=(_Float16)b2.y;  bf[10]=(_Float16)b2.z; bf[11]=(_Float16)b2.w;
    bf[12]=(_Float16)b3.x; bf[13]=(_Float16)b3.y; bf[14]=(_Float16)b3.z; bf[15]=(_Float16)b3.w;
    acc = __builtin_amdgcn_wmma_f32_16x16x32_f16(false, af, false, bf,
                                                 (short)0, acc, false, false);
  }
  const int mo = (lane >> 4) << 3;
#pragma unroll
  for (int r = 0; r < 8; ++r)
    scores[(((size_t)b * NHEAD + hh) * SEQ + mo + r) * SEQ + (lane & 15)] = acc[r];
}

// -------- softmax with causal mask + 1/sqrt(HD) scale, in-place ------------
__global__ __launch_bounds__(256) void gpt_softmax(float* __restrict__ scores) {
  const int idx = blockIdx.x * blockDim.x + threadIdx.x;   // B*H*T rows
  if (idx >= BATCH * NHEAD * SEQ) return;
  const int t = idx & (SEQ - 1);
  float* row = scores + (size_t)idx * SEQ;
  float vals[SEQ];
  float mx = -3.4e38f;
  for (int s = 0; s <= t; ++s) { vals[s] = row[s] * 0.125f; mx = fmaxf(mx, vals[s]); }
  float sum = 0.0f;
  for (int s = 0; s <= t; ++s) { vals[s] = expf(vals[s] - mx); sum += vals[s]; }
  const float inv = 1.0f / sum;
  for (int s = 0; s < SEQ; ++s) row[s] = (s <= t) ? vals[s] * inv : 0.0f;
}

// -------- O = attn @ V (per b,h; HD split into 4 N-tiles; K padded 16->32) -
__global__ __launch_bounds__(64) void gpt_attn_av(
    const float* __restrict__ attn, const float* __restrict__ v,
    float* __restrict__ o) {
  const int lane = threadIdx.x & 31;
  const int gid  = blockIdx.x * 2 + (threadIdx.x >> 5);    // B*H*4 waves
  const int nt   = gid & 3;
  const int bh   = gid >> 2;
  const int b    = bh >> 3;
  const int hh   = bh & 7;
  const int mn   = lane & 15;
  const int ka   = (lane >> 4) << 3;
  const int kb   = (lane >> 4) << 4;

  const float* pa = attn + (((size_t)b * NHEAD + hh) * SEQ + mn) * SEQ + ka;
  float4 a0 = *(const float4*)(pa);
  float4 a1 = *(const float4*)(pa + 4);
  v16h af;
  af[0]=(_Float16)a0.x; af[1]=(_Float16)a0.y; af[2]=(_Float16)a0.z; af[3]=(_Float16)a0.w;
  af[4]=(_Float16)a1.x; af[5]=(_Float16)a1.y; af[6]=(_Float16)a1.z; af[7]=(_Float16)a1.w;
#pragma unroll
  for (int j = 8; j < 16; ++j) af[j] = (_Float16)0.0f;

  v16h bf;
#pragma unroll
  for (int j = 0; j < 16; ++j) {
    const int kd = (j < 8) ? (ka + j) : (16 + ka + (j - 8));
    bf[j] = (kd < SEQ)
        ? (_Float16)v[(((size_t)b * SEQ + kd) * NHEAD + hh) * HEADD + nt * 16 + mn]
        : (_Float16)0.0f;
  }
  v8f acc = {};
  acc = __builtin_amdgcn_wmma_f32_16x16x32_f16(false, af, false, bf,
                                               (short)0, acc, false, false);
  const int mo = (lane >> 4) << 3;
#pragma unroll
  for (int r = 0; r < 8; ++r)
    o[(((size_t)b * SEQ + mo + r) * NHEAD + hh) * HEADD + nt * 16 + mn] = acc[r];
}

// -------- LayerNorm (one wave32 per row, shfl_xor reduction) ---------------
// out = (resid ? resid + LN(x) : LN(x))
__global__ __launch_bounds__(256) void gpt_layernorm(
    const float* __restrict__ x, const float* __restrict__ sc,
    const float* __restrict__ bi, const float* __restrict__ resid,
    float* __restrict__ out, int rows) {
  const int lane = threadIdx.x & 31;
  const int row  = blockIdx.x * 8 + (threadIdx.x >> 5);
  if (row >= rows) return;
  const float* p = x + (size_t)row * DMODEL;
  float s1 = 0.0f, s2 = 0.0f;
  for (int d = lane; d < DMODEL; d += 32) { float v = p[d]; s1 += v; s2 += v * v; }
#pragma unroll
  for (int off = 16; off > 0; off >>= 1) {
    s1 += __shfl_xor(s1, off, 32);
    s2 += __shfl_xor(s2, off, 32);
  }
  const float mu   = s1 * (1.0f / DMODEL);
  const float var  = s2 * (1.0f / DMODEL) - mu * mu;
  const float rstd = rsqrtf(var + 1e-5f);
  for (int d = lane; d < DMODEL; d += 32) {
    float v = (p[d] - mu) * rstd * sc[d] + bi[d];
    if (resid) v += resid[(size_t)row * DMODEL + d];
    out[(size_t)row * DMODEL + d] = v;
  }
}

// -------- Embedding gather + sinusoidal positional encoding ----------------
__global__ __launch_bounds__(256) void gpt_embed(
    const int* __restrict__ tok, const float* __restrict__ emb,
    float* __restrict__ x) {
  const int idx = blockIdx.x * blockDim.x + threadIdx.x;
  if (idx >= MROWS * DMODEL) return;
  const int d  = idx & (DMODEL - 1);
  const int bt = idx >> 9;
  const int t  = bt & (SEQ - 1);
  const int token = tok[bt];
  const int i2 = d >> 1;
  const float ang = (float)t * expf(-logf(10000.0f) * (float)(2 * i2) / (float)DMODEL);
  const float pe  = (d & 1) ? cosf(ang) : sinf(ang);
  x[idx] = emb[(size_t)token * DMODEL + d] + pe;
}

// ---------------------------------------------------------------------------
extern "C" void kernel_launch(void* const* d_in, const int* in_sizes, int n_in,
                              void* d_out, int out_size, void* d_ws, size_t ws_size,
                              hipStream_t stream) {
  (void)in_sizes; (void)n_in; (void)out_size; (void)ws_size;

  const int*   tok   = (const int*)  d_in[0];
  const float* emb   = (const float*)d_in[1];
  const float* Wq    = (const float*)d_in[2];
  const float* Wk    = (const float*)d_in[3];
  const float* Wv    = (const float*)d_in[4];
  const float* Wo    = (const float*)d_in[5];
  const float* bo    = (const float*)d_in[6];
  const float* ln1s  = (const float*)d_in[7];
  const float* ln1b  = (const float*)d_in[8];
  const float* ln2s  = (const float*)d_in[9];
  const float* ln2b  = (const float*)d_in[10];
  const float* W1    = (const float*)d_in[11];
  const float* b1    = (const float*)d_in[12];
  const float* W2    = (const float*)d_in[13];
  const float* b2    = (const float*)d_in[14];
  const float* lnfs  = (const float*)d_in[15];
  const float* lnfb  = (const float*)d_in[16];
  const float* Wvoc  = (const float*)d_in[17];
  const float* bvoc  = (const float*)d_in[18];
  float* out = (float*)d_out;

  // workspace layout (floats); total ~6.43M floats (~25.7 MB)
  float* ws  = (float*)d_ws;
  float* x   = ws;                         // [1024,512]
  float* x2  = ws + 524288;                // [1024,512]
  float* h   = ws + 1048576;               // [1024,512]  LN output
  float* q   = ws + 1572864;               // [1024,512]
  float* kbf = ws + 2097152;               // [1024,512]
  float* vbf = ws + 2621440;               // [1024,512]
  float* o   = ws + 3145728;               // [1024,512]
  float* scr = ws + 3670016;               // [64,8,16,16]
  float* ffh = ws + 3801088;               // [1024,2048]
  float* f   = ws + 5898240;               // [1024,512]

  gpt_embed<<<(MROWS * DMODEL) / 256, 256, 0, stream>>>(tok, emb, x);

  const size_t wqk_stride = (size_t)DMODEL * DMODEL;       // 512*512 per layer
  const size_t wff_stride = (size_t)DMODEL * FFDIM;        // 512*2048 per layer

  // GEMM grids: (M/64) * (N/32) blocks of 256 threads
  const int gQKV = (MROWS / 64) * (DMODEL / 32);           // 16*16  = 256
  const int gFF1 = (MROWS / 64) * (FFDIM / 32);            // 16*64  = 1024
  const int gVOC = (MROWS / 64) * (VOCAB / 32);            // 16*3133 = 50128

  for (int l = 0; l < LAYERS; ++l) {
    // h = LN1(x)
    gpt_layernorm<<<MROWS / 8, 256, 0, stream>>>(x, ln1s + l * DMODEL, ln1b + l * DMODEL,
                                                 nullptr, h, MROWS);
    // q,k,v = h @ W{q,k,v}[l]
    gpt_gemm_wmma<<<gQKV, 256, 0, stream>>>(h, Wq + l * wqk_stride, nullptr, nullptr,
                                            q,   MROWS, DMODEL, DMODEL, 0);
    gpt_gemm_wmma<<<gQKV, 256, 0, stream>>>(h, Wk + l * wqk_stride, nullptr, nullptr,
                                            kbf, MROWS, DMODEL, DMODEL, 0);
    gpt_gemm_wmma<<<gQKV, 256, 0, stream>>>(h, Wv + l * wqk_stride, nullptr, nullptr,
                                            vbf, MROWS, DMODEL, DMODEL, 0);
    // attention
    gpt_attn_scores<<<(BATCH * NHEAD) / 2, 64, 0, stream>>>(q, kbf, scr);
    gpt_softmax<<<(BATCH * NHEAD * SEQ) / 256, 256, 0, stream>>>(scr);
    gpt_attn_av<<<(BATCH * NHEAD * 4) / 2, 64, 0, stream>>>(scr, vbf, o);
    // x2 = x + o @ Wo[l] + bo[l]
    gpt_gemm_wmma<<<gQKV, 256, 0, stream>>>(o, Wo + l * wqk_stride, bo + l * DMODEL, x,
                                            x2, MROWS, DMODEL, DMODEL, 3);
    // ffh = relu(x2 @ W1[l] + b1[l])
    gpt_gemm_wmma<<<gFF1, 256, 0, stream>>>(x2, W1 + l * wff_stride, b1 + l * FFDIM, nullptr,
                                            ffh, MROWS, FFDIM, DMODEL, 2);
    // f = ffh @ W2[l] + b2[l]
    gpt_gemm_wmma<<<gQKV, 256, 0, stream>>>(ffh, W2 + l * wff_stride, b2 + l * DMODEL, nullptr,
                                            f, MROWS, DMODEL, FFDIM, 1);
    // x = x2 + LN2(f)
    gpt_layernorm<<<MROWS / 8, 256, 0, stream>>>(f, ln2s + l * DMODEL, ln2b + l * DMODEL,
                                                 x2, x, MROWS);
  }

  // h = LN_f(x);  out = h @ Wvoc + bvoc
  gpt_layernorm<<<MROWS / 8, 256, 0, stream>>>(x, lnfs, lnfb, nullptr, h, MROWS);
  gpt_gemm_wmma<<<gVOC, 256, 0, stream>>>(h, Wvoc, bvoc, nullptr, out,
                                          MROWS, VOCAB, DMODEL, 1);
}
